// SDFL_15805479649516
// MI455X (gfx1250) — matrix operand order repaired and checked by reading, not compiled, not run on hardware
//
#include <hip/hip_runtime.h>

typedef __attribute__((ext_vector_type(16))) _Float16 v16h;
typedef __attribute__((ext_vector_type(8)))  float    v8f;

#define WMMA_F16(a, b, c) \
  __builtin_amdgcn_wmma_f32_16x16x32_f16(false, (a), false, (b), (short)0, (c), false, false)

// ---- problem constants (from the reference) ----
constexpr int Bb   = 8;
constexpr int Hh   = 128;
constexpr int Ww   = 128;
constexpr int Nn   = Hh * Ww;      // 16384
constexpr int DIM  = 256;
constexpr int NHh  = 4;
constexpr int HDh  = 32;           // DIM/2/NH
constexpr int CHh  = 128;          // DIM/2
constexpr int D_IN = 23;           // #inv-freqs surviving the 2*pi/MAXSEQ cutoff
constexpr int D_OUT= 16;           // 2*HD//4

// ============ CDNA5 async global->LDS staging (ASYNCcnt tracked) ============
// Copies `bytes` (multiple of 4096, block of 256 threads) from contiguous
// global memory into the kernel's (single) static LDS allocation at offset 0.
// Each lane issues GLOBAL_LOAD_ASYNC_TO_LDS_B128 (16B per lane per issue),
// then the wave drains its ASYNCcnt and the block barriers.
__device__ __forceinline__ void async_stage_to_lds(const void* gsrc, int bytes) {
  unsigned long long g = (unsigned long long)(uintptr_t)gsrc +
                         (unsigned long long)(threadIdx.x * 16);
  unsigned l = threadIdx.x * 16;   // LDS byte offset (single shared array @ 0)
#pragma unroll
  for (int it = 0; it < 4 * 4096; it += 4096) {
    if (it < bytes)
      asm volatile("global_load_async_to_lds_b128 %0, %1, off"
                   :: "v"(l + (unsigned)it), "v"(g + (unsigned long long)it)
                   : "memory");
  }
  asm volatile("s_wait_asynccnt 0x0" ::: "memory");
  __syncthreads();
}

// ================= WMMA tile loaders (CDNA5 wave32 layouts) =================

// A-matrix 16x32 f16 layout from a row-major (M x K) f32 tile.
// lane L<16: M=L, K in {0..7, 16..23}; lane L>=16: M=L-16, K in {8..15, 24..31}
__device__ __forceinline__ v16h load_a_f32(const float* __restrict__ base, int ld, int lane) {
  int r  = lane & 15;
  int kb = (lane < 16) ? 0 : 8;
  const float* p = base + (size_t)r * ld + kb;
  v16h x;
#pragma unroll
  for (int i = 0; i < 8; ++i) x[i] = (_Float16)p[i];
#pragma unroll
  for (int i = 0; i < 8; ++i) x[8 + i] = (_Float16)p[16 + i];
  return x;
}

__device__ __forceinline__ v16h load_a_f16(const _Float16* __restrict__ base, int ld, int lane) {
  int r  = lane & 15;
  int kb = (lane < 16) ? 0 : 8;
  const _Float16* p = base + (size_t)r * ld + kb;
  v16h x;
#pragma unroll
  for (int i = 0; i < 8; ++i) x[i] = p[i];
#pragma unroll
  for (int i = 0; i < 8; ++i) x[8 + i] = p[16 + i];
  return x;
}

// B-matrix 32x16 f16 layout from a row-major (K x N) f32 tile.
// lanes 0-15: col=lane, K in {0..7, 16..23}; lanes 16-31: col=lane-16, K in {8..15, 24..31}
__device__ __forceinline__ v16h load_b_f32(const float* __restrict__ base, int ld, int lane) {
  int c  = lane & 15;
  int kb = (lane < 16) ? 0 : 8;
  v16h x;
#pragma unroll
  for (int i = 0; i < 16; ++i) x[i] = (_Float16)base[(size_t)(kb + i + ((i >= 8) ? 8 : 0)) * ld + c];
  return x;
}

// B-matrix from a row-major (N x K) tile (weights stored W[n,k]; K contiguous)
__device__ __forceinline__ v16h load_bt_f32(const float* __restrict__ base, int ld, int lane) {
  int c  = lane & 15;
  int kb = (lane < 16) ? 0 : 8;
  const float* p = base + (size_t)c * ld + kb;
  v16h x;
#pragma unroll
  for (int i = 0; i < 16; ++i) x[i] = (_Float16)p[i + ((i >= 8) ? 8 : 0)];
  return x;
}

// FOPE rotary: out[d] = t[d]*fc[d] - rot(t)[d]*fs[d], rot = [-t[32:], t[:32]]
__device__ __forceinline__ float fope_val(const _Float16* __restrict__ trow,
                                          const float* __restrict__ fcrow,
                                          const float* __restrict__ fsrow, int d) {
  float t = (float)trow[d];
  float r = (d < HDh) ? -(float)trow[d + HDh] : (float)trow[d - HDh];
  return t * fcrow[d] - r * fsrow[d];
}

// ====================== kernel 1: q/k/v = x @ W^T + b =======================
// M=131072, K=256, Nout=128.  grid.x = M/16, block = 256 (8 waves = 8 col tiles)
// A tile (16 x 256 f32 = 16 KB, contiguous) is async-staged to LDS once and
// shared by all 8 waves instead of 8 redundant global reads.
__global__ void qkv_kernel(const float* __restrict__ X, const float* __restrict__ W,
                           const float* __restrict__ bias, float* __restrict__ Out) {
  __shared__ float sX[16 * DIM];                   // 16 KB, LDS offset 0
  int mt = blockIdx.x;
  int wave = threadIdx.x >> 5, lane = threadIdx.x & 31;
  int nt = wave;                                   // 8 waves cover 128 cols
  async_stage_to_lds(X + (size_t)mt * 16 * DIM, 16 * DIM * 4);
  const float* wp = W + (size_t)nt * 16 * DIM;
  v8f acc = {};
  for (int k0 = 0; k0 < DIM; k0 += 32) {
    v16h a  = load_a_f32(sX + k0, DIM, lane);      // ds reads
    v16h bm = load_bt_f32(wp + k0, DIM, lane);
    acc = WMMA_F16(a, bm, acc);
  }
  int col = lane & 15, mb = (lane < 16) ? 0 : 8;
  float bb = bias[nt * 16 + col];
#pragma unroll
  for (int i = 0; i < 8; ++i)
    Out[((size_t)mt * 16 + mb + i) * CHh + nt * 16 + col] = acc[i] + bb;
}

// ================== kernel 2: FOPE sin/cos tables (NH,N,64) =================
__global__ void fope_kernel(const float* __restrict__ sinc, const float* __restrict__ cosc,
                            float* __restrict__ fs, float* __restrict__ fc) {
  int t = blockIdx.x * blockDim.x + threadIdx.x;   // token position
  int h = blockIdx.y;
  if (t >= Nn) return;
  float sv[D_IN], cv[D_IN];
#pragma unroll
  for (int D = 0; D < D_IN; ++D) {
    // inv_freq[D] = 10000^(-D/32)
    float inv = __expf(-(float)D * 0.28782313662425572f);  // ln(10000)/32
    float ang = (float)t * inv;
    sv[D] = __sinf(ang);
    cv[D] = __cosf(ang);
  }
  float fsd[HDh], fcd[HDh];
  for (int d = 0; d < D_OUT; ++d) {
    float ssum = 0.f, csum = 0.f, sacc = 0.f, cacc = 0.f;
    for (int D = 0; D < D_IN; ++D) {
      float sc = sinc[((size_t)h * D_IN + D) * D_OUT + d];
      float cc = cosc[((size_t)h * D_IN + D) * D_OUT + d];
      ssum += sc; csum += cc;
      sacc += sv[D] * sc; cacc += cv[D] * cc;
    }
    fsd[d] = sacc / ssum;
    fcd[d] = cacc / csum;
  }
#pragma unroll
  for (int d = D_OUT; d < HDh; ++d) { fsd[d] = 1.f; fcd[d] = 1.f; }   // pad = 1.0
  float* fsp = fs + ((size_t)h * Nn + t) * 64;
  float* fcp = fc + ((size_t)h * Nn + t) * 64;
#pragma unroll
  for (int d = 0; d < HDh; ++d) {
    fsp[d] = fsd[d]; fsp[HDh + d] = fsd[d];
    fcp[d] = fcd[d]; fcp[HDh + d] = fcd[d];
  }
}

// ============ kernel 3: hedgehog feature map -> f16 (B,NH,N,64) =============
__global__ void hedgehog_kernel(const float* __restrict__ src, const float* __restrict__ Wh,
                                const float* __restrict__ bh, _Float16* __restrict__ dst) {
  __shared__ float sW[HDh * HDh];
  __shared__ float sb[HDh];
  for (int i = threadIdx.x; i < HDh * HDh; i += blockDim.x) sW[i] = Wh[i];
  if (threadIdx.x < HDh) sb[threadIdx.x] = bh[threadIdx.x];
  __syncthreads();
  size_t gid = (size_t)blockIdx.x * blockDim.x + threadIdx.x;   // (b*NH+h)*N + n
  int n  = (int)(gid % Nn);
  int hh = (int)((gid / Nn) % NHh);
  int b  = (int)(gid / ((size_t)Nn * NHh));
  const float* tp = src + ((size_t)b * Nn + n) * CHh + hh * HDh;
  float td[HDh];
#pragma unroll
  for (int d = 0; d < HDh; ++d) td[d] = tp[d];
  float y[HDh];
  float mx = -1e30f;
  for (int e = 0; e < HDh; ++e) {
    float acc = sb[e];
#pragma unroll
    for (int d = 0; d < HDh; ++d) acc += td[d] * sW[e * HDh + d];
    y[e] = 2.f * acc;
    mx = fmaxf(mx, fabsf(y[e]));
  }
  float sum = 0.f;
  for (int e = 0; e < HDh; ++e) sum += __expf(y[e] - mx) + __expf(-y[e] - mx);
  float inv = 1.f / sum;
  _Float16* op = dst + gid * 64;
  for (int e = 0; e < HDh; ++e) {
    op[e]       = (_Float16)(__expf(y[e] - mx) * inv);
    op[HDh + e] = (_Float16)(__expf(-y[e] - mx) * inv);
  }
}

// =================== kernel 4: kmean[b,h,d] = mean_n kf =====================
__global__ void kmean_kernel(const _Float16* __restrict__ kf, float* __restrict__ km) {
  __shared__ float red[256];
  int bh_ = blockIdx.x;
  int d = threadIdx.x & 63, sl = threadIdx.x >> 6;
  const _Float16* p = kf + (size_t)bh_ * Nn * 64 + d;
  float acc = 0.f;
  for (int n = sl; n < Nn; n += 4) acc += (float)p[(size_t)n * 64];
  red[threadIdx.x] = acc;
  __syncthreads();
  if (sl == 0)
    km[bh_ * 64 + d] = (red[d] + red[64 + d] + red[128 + d] + red[192 + d]) * (1.f / Nn);
}

// ============ kernel 5: kv[b,h,d,e] = (1/N) sum_n kfope[n,d] v[n,e] =========
// grid = 32 (b,h), block = 256: wave -> (d-tile 0..3, e-tile 0..1), K = 16384
__global__ void kv_kernel(const _Float16* __restrict__ kf, const float* __restrict__ fs,
                          const float* __restrict__ fc, const float* __restrict__ v,
                          float* __restrict__ kvb) {
  int bh_ = blockIdx.x, b = bh_ >> 2, h = bh_ & 3;
  int wave = threadIdx.x >> 5, lane = threadIdx.x & 31;
  int dt = wave >> 1, et = wave & 1;
  const _Float16* kfp = kf + (size_t)bh_ * Nn * 64;
  const float* fsp = fs + (size_t)h * Nn * 64;
  const float* fcp = fc + (size_t)h * Nn * 64;
  const float* vp  = v + (size_t)b * Nn * CHh + h * HDh + et * 16;
  int drow = dt * 16 + (lane & 15);
  int kbA = (lane < 16) ? 0 : 8;
  v8f acc = {};
  for (int n0 = 0; n0 < Nn; n0 += 32) {
    // prefetch the next chunk (global_prefetch_b8) while this WMMA streams
    __builtin_prefetch(kfp + (size_t)(n0 + 64) * 64, 0, 0);
    __builtin_prefetch(vp + (size_t)(n0 + 64) * CHh, 0, 0);
    v16h a;
#pragma unroll
    for (int i = 0; i < 16; ++i) {
      int n = n0 + kbA + i + ((i >= 8) ? 8 : 0);
      a[i] = (_Float16)fope_val(kfp + (size_t)n * 64, fcp + (size_t)n * 64,
                                fsp + (size_t)n * 64, drow);
    }
    v16h bm = load_b_f32(vp + (size_t)n0 * CHh, CHh, lane);
    acc = WMMA_F16(a, bm, acc);
  }
  int col = lane & 15, mb = (lane < 16) ? 0 : 8;
#pragma unroll
  for (int i = 0; i < 8; ++i)
    kvb[(size_t)bh_ * 2048 + (size_t)(dt * 16 + mb + i) * 32 + et * 16 + col] =
        acc[i] * (1.f / Nn);
}

// ================= kernel 6: z = 1 / (qf . kmean + 1e-6) ====================
__global__ void z_kernel(const _Float16* __restrict__ qf, const float* __restrict__ km,
                         float* __restrict__ z) {
  size_t gid = (size_t)blockIdx.x * blockDim.x + threadIdx.x;
  const _Float16* p = qf + gid * 64;
  const float* kmp = km + (gid / Nn) * 64;
  float acc = 1e-6f;
#pragma unroll
  for (int d = 0; d < 64; ++d) acc += (float)p[d] * kmp[d];
  z[gid] = 1.f / acc;
}

// ===== kernel 7: xs = (q_fope @ kv) * z  -> cat[..., 0:128] (f16) ===========
// grid = (N/16, 32), block = 64 (2 waves = 2 e-tiles), K = 64
__global__ void xs_kernel(const _Float16* __restrict__ qf, const float* __restrict__ fs,
                          const float* __restrict__ fc, const float* __restrict__ kvb,
                          const float* __restrict__ z, _Float16* __restrict__ cat) {
  int bh_ = blockIdx.y, b = bh_ >> 2, h = bh_ & 3;
  int mt = blockIdx.x;
  int wave = threadIdx.x >> 5, lane = threadIdx.x & 31;    // wave = e-tile
  const _Float16* qfp = qf + ((size_t)bh_ * Nn + mt * 16) * 64;
  const float* fsp = fs + ((size_t)h * Nn + mt * 16) * 64;
  const float* fcp = fc + ((size_t)h * Nn + mt * 16) * 64;
  const float* kvp = kvb + (size_t)bh_ * 2048 + wave * 16;
  int row = lane & 15, kbA = (lane < 16) ? 0 : 8;
  const _Float16* qrow = qfp + (size_t)row * 64;
  const float* fcrow = fcp + (size_t)row * 64;
  const float* fsrow = fsp + (size_t)row * 64;
  v8f acc = {};
  for (int k0 = 0; k0 < 64; k0 += 32) {
    v16h a;
#pragma unroll
    for (int i = 0; i < 16; ++i) {
      int kk = kbA + i + ((i >= 8) ? 8 : 0);
      a[i] = (_Float16)fope_val(qrow, fcrow, fsrow, k0 + kk);
    }
    v16h bm = load_b_f32(kvp + (size_t)k0 * 32, 32, lane);
    acc = WMMA_F16(a, bm, acc);
  }
  int col = lane & 15, mb = (lane < 16) ? 0 : 8;
#pragma unroll
  for (int i = 0; i < 8; ++i) {
    int m = mt * 16 + mb + i;
    float zv = z[(size_t)bh_ * Nn + m];
    cat[((size_t)b * Nn + m) * DIM + h * HDh + wave * 16 + col] = (_Float16)(acc[i] * zv);
  }
}

// ========== kernel 8: depthwise 3x3 LEPE conv, added into cat low half ======
__global__ void lepe_kernel(const float* __restrict__ v, const float* __restrict__ lw,
                            const float* __restrict__ lb, _Float16* __restrict__ cat) {
  size_t gid = (size_t)blockIdx.x * blockDim.x + threadIdx.x;
  int c = (int)(gid % CHh);
  size_t r = gid / CHh;
  int xx = (int)(r % Ww); r /= Ww;
  int yy = (int)(r % Hh);
  int b  = (int)(r / Hh);
  float acc = lb[c];
#pragma unroll
  for (int ky = -1; ky <= 1; ++ky)
#pragma unroll
    for (int kx = -1; kx <= 1; ++kx) {
      int y2 = yy + ky, x2 = xx + kx;
      if (y2 >= 0 && y2 < Hh && x2 >= 0 && x2 < Ww)
        acc += lw[c * 9 + (ky + 1) * 3 + (kx + 1)] *
               v[((size_t)b * Nn + y2 * Ww + x2) * CHh + c];
    }
  size_t o = ((size_t)b * Nn + yy * Ww + xx) * DIM + c;
  cat[o] = (_Float16)((float)cat[o] + acc);
}

// ============== kernel 9: 1/max(||column||, 1e-12) over N ===================
__global__ void colnorm_kernel(const float* __restrict__ q, float* __restrict__ invn) {
  __shared__ float red[256];
  int bc = blockIdx.x, b = bc >> 7, c = bc & 127;
  const float* p = q + (size_t)b * Nn * CHh + c;
  float acc = 0.f;
  for (int n = threadIdx.x; n < Nn; n += 256) {
    float t = p[(size_t)n * CHh];
    acc += t * t;
  }
  red[threadIdx.x] = acc;
  __syncthreads();
  for (int s = 128; s > 0; s >>= 1) {
    if ((int)threadIdx.x < s) red[threadIdx.x] += red[threadIdx.x + s];
    __syncthreads();
  }
  if (threadIdx.x == 0) invn[bc] = 1.f / fmaxf(sqrtf(red[0]), 1e-12f);
}

// ============== kernel 10: Gram matrix G[b,c,d] = sum_l q[l,c] k[l,d] =======
// grid = (B, 64 tiles), block = 256: 8 waves split K=16384, LDS reduce
__global__ void gram_kernel(const float* __restrict__ q, const float* __restrict__ k,
                            float* __restrict__ G) {
  int b = blockIdx.x, tile = blockIdx.y, ct = tile >> 3, dt = tile & 7;
  int wave = threadIdx.x >> 5, lane = threadIdx.x & 31;
  const float* qp = q + (size_t)b * Nn * CHh + ct * 16;
  const float* kp = k + (size_t)b * Nn * CHh + dt * 16;
  int row = lane & 15, kbA = (lane < 16) ? 0 : 8;
  v8f acc = {};
  int l0b = wave * (Nn / 8), l0e = l0b + (Nn / 8);
  for (int l0 = l0b; l0 < l0e; l0 += 32) {
    __builtin_prefetch(qp + (size_t)(l0 + 64) * CHh, 0, 0);
    __builtin_prefetch(kp + (size_t)(l0 + 64) * CHh, 0, 0);
    v16h a;
#pragma unroll
    for (int i = 0; i < 16; ++i) {
      int kk = kbA + i + ((i >= 8) ? 8 : 0);
      a[i] = (_Float16)qp[(size_t)(l0 + kk) * CHh + row];
    }
    v16h bm = load_b_f32(kp + (size_t)l0 * CHh, CHh, lane);
    acc = WMMA_F16(a, bm, acc);
  }
  __shared__ float sred[8][256];
  int col = lane & 15, mb = (lane < 16) ? 0 : 8;
#pragma unroll
  for (int i = 0; i < 8; ++i) sred[wave][(mb + i) * 16 + col] = acc[i];
  __syncthreads();
  int m = threadIdx.x >> 4, n2 = threadIdx.x & 15;
  float s = 0.f;
#pragma unroll
  for (int w2 = 0; w2 < 8; ++w2) s += sred[w2][m * 16 + n2];
  G[((size_t)b * CHh + ct * 16 + m) * CHh + dt * 16 + n2] = s;
}

// =========== kernel 11: channel softmax with temperature + norms ============
__global__ void chsoft_kernel(const float* __restrict__ G, const float* __restrict__ invq,
                              const float* __restrict__ invk, const float* __restrict__ temp,
                              float* __restrict__ attn) {
  int bc = blockIdx.x, b = bc >> 7, c = bc & 127;
  int d = threadIdx.x;
  float logit = G[(size_t)bc * CHh + d] * invq[bc] * invk[(b << 7) + d] * temp[c];
  __shared__ float sh[128];
  sh[d] = logit;
  __syncthreads();
  for (int s = 64; s > 0; s >>= 1) {
    if (d < s) sh[d] = fmaxf(sh[d], sh[d + s]);
    __syncthreads();
  }
  float mx = sh[0];
  __syncthreads();
  float e = __expf(logit - mx);
  sh[d] = e;
  __syncthreads();
  for (int s = 64; s > 0; s >>= 1) {
    if (d < s) sh[d] += sh[d + s];
    __syncthreads();
  }
  attn[(size_t)bc * CHh + d] = e / sh[0];
}

// ====== kernel 12: x_channel[b,l,c] = sum_d attn[b,c,d] v[b,l,d] ============
// grid = (N/16, B), block = 256 (8 waves = 8 col tiles), K = 128
// A tile (16 x 128 f32 = 8 KB, contiguous) async-staged to LDS, shared by waves
__global__ void xch_kernel(const float* __restrict__ v, const float* __restrict__ attn,
                           _Float16* __restrict__ cat) {
  __shared__ float sV[16 * CHh];                   // 8 KB, LDS offset 0
  int b = blockIdx.y, lt = blockIdx.x;
  int wave = threadIdx.x >> 5, lane = threadIdx.x & 31;
  int ct = wave;
  async_stage_to_lds(v + ((size_t)b * Nn + lt * 16) * CHh, 16 * CHh * 4);
  const float* ap = attn + ((size_t)b * CHh + ct * 16) * CHh;   // (N x K) row-major
  v8f acc = {};
  for (int k0 = 0; k0 < CHh; k0 += 32) {
    v16h a  = load_a_f32(sV + k0, CHh, lane);
    v16h bm = load_bt_f32(ap + k0, CHh, lane);
    acc = WMMA_F16(a, bm, acc);
  }
  int col = lane & 15, mb = (lane < 16) ? 0 : 8;
#pragma unroll
  for (int i = 0; i < 8; ++i) {
    int l = lt * 16 + mb + i;
    cat[((size_t)b * Nn + l) * DIM + CHh + ct * 16 + col] = (_Float16)acc[i];
  }
}

// ============ kernel 13: out = cat @ Wp^T + bp  (M=131072, K=N=256) =========
// A tile (16 x 256 f16 = 8 KB, contiguous) async-staged to LDS, shared by waves
__global__ void proj_kernel(const _Float16* __restrict__ cat, const float* __restrict__ Wp,
                            const float* __restrict__ bp, float* __restrict__ out) {
  __shared__ _Float16 sA[16 * DIM];                // 8 KB, LDS offset 0
  int mt = blockIdx.x;
  int wave = threadIdx.x >> 5, lane = threadIdx.x & 31;
  async_stage_to_lds(cat + (size_t)mt * 16 * DIM, 16 * DIM * 2);
#pragma unroll
  for (int t = 0; t < 2; ++t) {
    int nt = wave * 2 + t;
    const float* wpp = Wp + (size_t)nt * 16 * DIM;
    v8f acc = {};
    for (int k0 = 0; k0 < DIM; k0 += 32) {
      v16h a  = load_a_f16(sA + k0, DIM, lane);
      v16h bm = load_bt_f32(wpp + k0, DIM, lane);
      acc = WMMA_F16(a, bm, acc);
    }
    int col = lane & 15, mb = (lane < 16) ? 0 : 8;
    float bb = bp[nt * 16 + col];
#pragma unroll
    for (int i = 0; i < 8; ++i)
      out[((size_t)mt * 16 + mb + i) * DIM + nt * 16 + col] = acc[i] + bb;
  }
}

// ============================== host launcher ===============================
extern "C" void kernel_launch(void* const* d_in, const int* in_sizes, int n_in,
                              void* d_out, int out_size, void* d_ws, size_t ws_size,
                              hipStream_t stream) {
  (void)in_sizes; (void)n_in; (void)out_size; (void)ws_size;
  const float* x    = (const float*)d_in[0];
  const float* Wq   = (const float*)d_in[1];
  const float* bq   = (const float*)d_in[2];
  const float* Wk   = (const float*)d_in[3];
  const float* bk   = (const float*)d_in[4];
  const float* Wv   = (const float*)d_in[5];
  const float* bv   = (const float*)d_in[6];
  const float* Whq  = (const float*)d_in[7];
  const float* bhq  = (const float*)d_in[8];
  const float* Whk  = (const float*)d_in[9];
  const float* bhk  = (const float*)d_in[10];
  const float* sinc = (const float*)d_in[11];
  const float* cosc = (const float*)d_in[12];
  const float* lw   = (const float*)d_in[13];
  const float* lb   = (const float*)d_in[14];
  const float* temp = (const float*)d_in[15];
  const float* Wp   = (const float*)d_in[16];
  const float* bp   = (const float*)d_in[17];

  float* out  = (float*)d_out;                         // (B,N,DIM)
  float* qbuf = out + (size_t)Bb * Nn * DIM;           // output q (B,N,CH)
  float* kbuf = qbuf + (size_t)Bb * Nn * CHh;          // output k (B,N,CH)

  char* ws = (char*)d_ws;
  size_t off = 0;
  float*    vbuf  = (float*)(ws + off);    off += (size_t)Bb * Nn * CHh * 4;       //  64 MiB
  _Float16* qf    = (_Float16*)(ws + off); off += (size_t)Bb * NHh * Nn * 64 * 2;  //  64 MiB
  _Float16* kf    = (_Float16*)(ws + off); off += (size_t)Bb * NHh * Nn * 64 * 2;  //  64 MiB
  float*    fs    = (float*)(ws + off);    off += (size_t)NHh * Nn * 64 * 4;       //  16 MiB
  float*    fc    = (float*)(ws + off);    off += (size_t)NHh * Nn * 64 * 4;       //  16 MiB
  float*    km    = (float*)(ws + off);    off += (size_t)Bb * NHh * 64 * 4;
  float*    kvb   = (float*)(ws + off);    off += (size_t)Bb * NHh * 64 * 32 * 4;
  float*    zb    = (float*)(ws + off);    off += (size_t)Bb * NHh * Nn * 4;       //   2 MiB
  float*    qnb   = (float*)(ws + off);    off += (size_t)Bb * CHh * 4;
  float*    knb   = (float*)(ws + off);    off += (size_t)Bb * CHh * 4;
  float*    Gb    = (float*)(ws + off);    off += (size_t)Bb * CHh * CHh * 4;
  float*    attnb = (float*)(ws + off);    off += (size_t)Bb * CHh * CHh * 4;
  _Float16* cat   = (_Float16*)(ws + off); off += (size_t)Bb * Nn * DIM * 2;       //  64 MiB

  const int MT = (Bb * Nn) / 16;   // 8192 row tiles

  // QKV projections (q,k also are outputs 2 and 3)
  qkv_kernel<<<MT, 256, 0, stream>>>(x, Wq, bq, qbuf);
  qkv_kernel<<<MT, 256, 0, stream>>>(x, Wk, bk, kbuf);
  qkv_kernel<<<MT, 256, 0, stream>>>(x, Wv, bv, vbuf);

  // FOPE tables + hedgehog features
  fope_kernel<<<dim3(Nn / 256, NHh), 256, 0, stream>>>(sinc, cosc, fs, fc);
  hedgehog_kernel<<<(Bb * NHh * Nn) / 256, 256, 0, stream>>>(qbuf, Whq, bhq, qf);
  hedgehog_kernel<<<(Bb * NHh * Nn) / 256, 256, 0, stream>>>(kbuf, Whk, bhk, kf);

  // linear attention
  kmean_kernel<<<Bb * NHh, 256, 0, stream>>>(kf, km);
  kv_kernel<<<Bb * NHh, 256, 0, stream>>>(kf, fs, fc, vbuf, kvb);
  z_kernel<<<(Bb * NHh * Nn) / 256, 256, 0, stream>>>(qf, km, zb);
  xs_kernel<<<dim3(Nn / 16, Bb * NHh), 64, 0, stream>>>(qf, fs, fc, kvb, zb, cat);
  lepe_kernel<<<(Bb * Nn * CHh) / 256, 256, 0, stream>>>(vbuf, lw, lb, cat);

  // channel attention
  colnorm_kernel<<<Bb * CHh, 256, 0, stream>>>(qbuf, qnb);
  colnorm_kernel<<<Bb * CHh, 256, 0, stream>>>(kbuf, knb);
  gram_kernel<<<dim3(Bb, 64), 256, 0, stream>>>(qbuf, kbuf, Gb);
  chsoft_kernel<<<Bb * CHh, 128, 0, stream>>>(Gb, qnb, knb, temp, attnb);
  xch_kernel<<<dim3(Nn / 16, Bb), 256, 0, stream>>>(vbuf, attnb, cat);

  // output projection
  proj_kernel<<<MT, 256, 0, stream>>>(cat, Wp, bp, out);
}
